// RWKV_46222438040171
// MI455X (gfx1250) — matrix operand order repaired and checked by reading, compiled, not run on hardware
//
#include <hip/hip_runtime.h>
#include <hip/hip_bf16.h>

// ---------------------------------------------------------------------------
// RWKV-7 forward, MI455X (gfx1250).
// GEMMs: v_wmma_f32_16x16x32_f16, async global->LDS staging, double buffer.
// Weights pre-transposed+padded to [Npad,K] f16 so staging is unguarded.
// ---------------------------------------------------------------------------

typedef __attribute__((ext_vector_type(16))) _Float16 v16h;
typedef __attribute__((ext_vector_type(8)))  _Float16 v8h;
typedef __attribute__((ext_vector_type(8)))  float    v8f;

#define C_DIM  1024
#define T_DIM  1024
#define B_DIM  2
#define BT     2048            // B*T
#define H_DIM  16
#define N_DIM  64
#define V_DIM  32000
#define F_DIM  4096
#define GN_EPS 6.4e-4f         // 1e-5 * 8^2

// ---------------------------- GEMM (WMMA) ----------------------------------
// C[M,Nout] = A[M,K] * Bt[Npad,K]^T ; A,Bt f16 row-major, C f32.
// Requires M % 128 == 0, K % 32 == 0, Npad = ceil(Nout/128)*128 (zero-padded).
#define BM 128
#define BN 128
#define BK 32
#define LDT 40                 // padded LDS row stride (halves): conflict-free

static __device__ inline v16h pack16(v8h lo, v8h hi) {
    v16h r;
#pragma unroll
    for (int i = 0; i < 8; ++i) { r[i] = lo[i]; r[i + 8] = hi[i]; }
    return r;
}

// low 32 bits of a generic pointer to LDS == wave-relative LDS byte address
static __device__ inline unsigned lds_off(const void* p) {
    return (unsigned)(unsigned long long)p;
}
// CDNA5 async DMA: 16B global -> LDS, tracked by ASYNCcnt
static __device__ inline void async_b128(unsigned lds, const _Float16* g) {
    asm volatile("global_load_async_to_lds_b128 %0, %1, off"
                 :: "v"(lds), "v"((unsigned long long)g) : "memory");
}
static __device__ inline void wait_async0() {
    asm volatile("s_wait_asynccnt 0x0" ::: "memory");
}

__global__ __launch_bounds__(256) void k_gemm_f16(
    const _Float16* __restrict__ A, const _Float16* __restrict__ Bt,
    float* __restrict__ Cm, int M, int N, int K)
{
    __shared__ _Float16 As[2][BM * LDT];
    __shared__ _Float16 Bs[2][BM * LDT];

    const int tid  = threadIdx.x;
    const int lane = tid & 31;
    const int wid  = tid >> 5;
    const int bm   = blockIdx.y * BM;
    const int bn   = blockIdx.x * BN;
    const int wm   = (wid & 3) * 32;   // 4 waves along M
    const int wn   = (wid >> 2) * 64;  // 2 waves along N

    // staging map: each thread copies 16 halves of one row (A) + one row (Bt)
    const int sr = tid >> 1;           // 0..127 row within tile
    const int sk = (tid & 1) * 16;     // 0/16 col group

    const _Float16* agp = A  + (size_t)(bm + sr) * K + sk;
    const _Float16* bgp = Bt + (size_t)(bn + sr) * K + sk;
    const unsigned la0 = lds_off(&As[0][sr * LDT + sk]);
    const unsigned lb0 = lds_off(&Bs[0][sr * LDT + sk]);
    const unsigned bufstride = (unsigned)(BM * LDT * sizeof(_Float16));

    v8f acc[2][4] = {};

    // prologue: stage k0 = 0 into buffer 0
    async_b128(la0,      agp);
    async_b128(la0 + 16, agp + 8);
    async_b128(lb0,      bgp);
    async_b128(lb0 + 16, bgp + 8);
    wait_async0();
    __syncthreads();

    int buf = 0;
    for (int k0 = 0; k0 < K; k0 += BK) {
        // stream next tile into the other buffer while computing
        if (k0 + BK < K) {
            const unsigned la = la0 + (buf ^ 1) * bufstride;
            const unsigned lb = lb0 + (buf ^ 1) * bufstride;
            const _Float16* ag = agp + k0 + BK;
            const _Float16* bg = bgp + k0 + BK;
            async_b128(la,      ag);
            async_b128(la + 16, ag + 8);
            async_b128(lb,      bg);
            async_b128(lb + 16, bg + 8);
        }

        // ---- fragments ----
        // A (16x32): lane m = lane&15; halves 0..7 -> K=kb.., 8..15 -> K=16+kb..
        const int am = lane & 15;
        const int kb = (lane >> 4) * 8;
        v16h afr[2];
#pragma unroll
        for (int mt = 0; mt < 2; ++mt) {
            const int m = wm + mt * 16 + am;
            v8h lo = *(const v8h*)&As[buf][m * LDT + kb];
            v8h hi = *(const v8h*)&As[buf][m * LDT + 16 + kb];
            afr[mt] = pack16(lo, hi);
        }
        // B (32x16): lane n = lane&15; halves 0..7 -> K=klo.., 8..15 -> K=16+klo..
        const int bnc = lane & 15;
        const int klo = (lane >> 4) * 8;
        v16h bfr[4];
#pragma unroll
        for (int nt = 0; nt < 4; ++nt) {
            const int n = wn + nt * 16 + bnc;
            v8h lo = *(const v8h*)&Bs[buf][n * LDT + klo];
            v8h hi = *(const v8h*)&Bs[buf][n * LDT + 16 + klo];
            bfr[nt] = pack16(lo, hi);
        }
#pragma unroll
        for (int mt = 0; mt < 2; ++mt)
#pragma unroll
            for (int nt = 0; nt < 4; ++nt)
                acc[mt][nt] = __builtin_amdgcn_wmma_f32_16x16x32_f16(
                    false, afr[mt], false, bfr[nt],
                    (short)0, acc[mt][nt], false, false);

        wait_async0();      // our async writes (next buffer) landed
        __syncthreads();    // everyone done reading current buffer
        buf ^= 1;
    }

    // ---- store C: vgpr r, lane l -> M = r + (l>>4)*8, N = l&15 ----
    const int lhi  = (lane >> 4) * 8;
    const int lcol = lane & 15;
    if (bn + BN <= N) {
#pragma unroll
        for (int mt = 0; mt < 2; ++mt)
#pragma unroll
            for (int nt = 0; nt < 4; ++nt)
#pragma unroll
                for (int rr = 0; rr < 8; ++rr) {
                    const int row = bm + wm + mt * 16 + rr + lhi;
                    const int col = bn + wn + nt * 16 + lcol;
                    Cm[(size_t)row * N + col] = acc[mt][nt][rr];
                }
    } else {
#pragma unroll
        for (int mt = 0; mt < 2; ++mt)
#pragma unroll
            for (int nt = 0; nt < 4; ++nt)
#pragma unroll
                for (int rr = 0; rr < 8; ++rr) {
                    const int row = bm + wm + mt * 16 + rr + lhi;
                    const int col = bn + wn + nt * 16 + lcol;
                    if (col < N) Cm[(size_t)row * N + col] = acc[mt][nt][rr];
                }
    }
}

// ---- transpose + convert + pad: src[K,N] f32 -> dst[Npad,K] f16 ------------
__global__ __launch_bounds__(256) void k_cvtT16(
    const float* __restrict__ src, _Float16* __restrict__ dst,
    int K, int N, int Npad)
{
    __shared__ float tile[32][33];
    const int kb = blockIdx.x * 32, nb = blockIdx.y * 32;
    const int tx = threadIdx.x, ty = threadIdx.y;      // 32 x 8
#pragma unroll
    for (int i = ty; i < 32; i += 8) {
        const int k = kb + i, n = nb + tx;
        tile[i][tx] = (n < N) ? src[(size_t)k * N + n] : 0.f;   // K % 32 == 0
    }
    __syncthreads();
#pragma unroll
    for (int i = ty; i < 32; i += 8) {
        const int n = nb + i, k = kb + tx;
        if (n < Npad) dst[(size_t)n * K + k] = (_Float16)tile[tx][i];
    }
}

// ---------------------------- elementwise ----------------------------------
static __device__ inline float sigf(float x) { return 1.f / (1.f + __expf(-x)); }

__global__ void k_embed(const int* __restrict__ idx,
                        const float* __restrict__ emb, float* __restrict__ out)
{
    size_t i = (size_t)blockIdx.x * 256 + threadIdx.x;
    if (i >= (size_t)BT * C_DIM) return;
    const int row = (int)(i >> 10), c = (int)(i & 1023);
    out[i] = emb[(size_t)idx[row] * C_DIM + c];
}

__global__ __launch_bounds__(256) void k_layernorm(
    const float* __restrict__ x, const float* __restrict__ w,
    const float* __restrict__ b, float* __restrict__ out,
    _Float16* __restrict__ out16)
{
    const int row = blockIdx.x, tid = threadIdx.x, c0 = tid * 4;
    __shared__ float ps[256], pq[256];
    __shared__ float smean, sinv;
    const float* xr = x + (size_t)row * C_DIM;
    float v[4], s = 0.f, q = 0.f;
#pragma unroll
    for (int i = 0; i < 4; ++i) { v[i] = xr[c0 + i]; s += v[i]; q += v[i] * v[i]; }
    ps[tid] = s; pq[tid] = q;
    __syncthreads();
    for (int st = 128; st > 0; st >>= 1) {
        if (tid < st) { ps[tid] += ps[tid + st]; pq[tid] += pq[tid + st]; }
        __syncthreads();
    }
    if (tid == 0) {
        const float m = ps[0] * (1.f / C_DIM);
        const float var = pq[0] * (1.f / C_DIM) - m * m;
        smean = m; sinv = rsqrtf(var + 1e-5f);
    }
    __syncthreads();
#pragma unroll
    for (int i = 0; i < 4; ++i) {
        const int c = c0 + i;
        const float o = (v[i] - smean) * sinv * w[c] + b[c];
        if (out)   out[(size_t)row * C_DIM + c] = o;
        if (out16) out16[(size_t)row * C_DIM + c] = (_Float16)o;
    }
}

__global__ void k_tmix_pre(const float* __restrict__ x,
    const float* xr_c, const float* xw_c, const float* xk_c,
    const float* xv_c, const float* xa_c, const float* xg_c,
    _Float16* xr, _Float16* xw, _Float16* xk,
    _Float16* xv, _Float16* xa, _Float16* xg)
{
    size_t i = (size_t)blockIdx.x * 256 + threadIdx.x;
    if (i >= (size_t)BT * C_DIM) return;
    const int row = (int)(i >> 10), c = (int)(i & 1023);
    const int t = row & (T_DIM - 1);
    const float xc = x[i];
    const float prev = (t == 0) ? 0.f : x[i - C_DIM];
    const float xx = prev - xc;
    xr[i] = (_Float16)(xc + xx * xr_c[c]);
    xw[i] = (_Float16)(xc + xx * xw_c[c]);
    xk[i] = (_Float16)(xc + xx * xk_c[c]);
    xv[i] = (_Float16)(xc + xx * xv_c[c]);
    xa[i] = (_Float16)(xc + xx * xa_c[c]);
    xg[i] = (_Float16)(xc + xx * xg_c[c]);
}

__global__ void k_cmix_pre(const float* __restrict__ x,
                           const float* __restrict__ kc, _Float16* __restrict__ kx)
{
    size_t i = (size_t)blockIdx.x * 256 + threadIdx.x;
    if (i >= (size_t)BT * C_DIM) return;
    const int row = (int)(i >> 10), c = (int)(i & 1023);
    const int t = row & (T_DIM - 1);
    const float xc = x[i];
    const float prev = (t == 0) ? 0.f : x[i - C_DIM];
    kx[i] = (_Float16)(xc + (prev - xc) * kc[c]);
}

__global__ void k_cvt16(const float* in, _Float16* out, size_t n) {
    size_t i = (size_t)blockIdx.x * 256 + threadIdx.x;
    if (i < n) out[i] = (_Float16)in[i];
}
__global__ void k_tanh16(const float* in, _Float16* out, size_t n) {
    size_t i = (size_t)blockIdx.x * 256 + threadIdx.x;
    if (i < n) out[i] = (_Float16)tanhf(in[i]);
}
__global__ void k_sig16(const float* in, _Float16* out, size_t n) {
    size_t i = (size_t)blockIdx.x * 256 + threadIdx.x;
    if (i < n) out[i] = (_Float16)sigf(in[i]);
}
__global__ void k_relu2_16(const float* in, _Float16* out, size_t n) {
    size_t i = (size_t)blockIdx.x * 256 + threadIdx.x;
    if (i < n) { float r = fmaxf(in[i], 0.f); out[i] = (_Float16)(r * r); }
}
__global__ void k_add(float* x, const float* d, size_t n) {
    size_t i = (size_t)blockIdx.x * 256 + threadIdx.x;
    if (i < n) x[i] += d[i];
}
// wd = exp(-exp(-softplus(-(w0+y)) - 0.5))
__global__ void k_wact(const float* y, const float* w0, float* wd, size_t n) {
    size_t i = (size_t)blockIdx.x * 256 + threadIdx.x;
    if (i < n) {
        const float u = w0[i & 1023] + y[i];
        const float w = -log1pf(__expf(-u)) - 0.5f;
        wd[i] = __expf(-__expf(w));
    }
}
__global__ void k_aact(const float* y, const float* a0, float* a, size_t n) {
    size_t i = (size_t)blockIdx.x * 256 + threadIdx.x;
    if (i < n) a[i] = sigf(a0[i & 1023] + y[i]);
}
__global__ void k_vmix(float* v, const float* vf, const float* y,
                       const float* v0, size_t n) {
    size_t i = (size_t)blockIdx.x * 256 + threadIdx.x;
    if (i < n) {
        const float s = sigf(v0[i & 1023] + y[i]);
        v[i] += (vf[i] - v[i]) * s;
    }
}

// kk = normalize_per_head(k*k_k); an=-kk; bp=kk*a; kscan=k*(1+(a-1)*k_a)
__global__ __launch_bounds__(256) void k_kkprep(
    const float* __restrict__ kraw, const float* __restrict__ a,
    const float* __restrict__ kkw, const float* __restrict__ kaw,
    float* __restrict__ an, float* __restrict__ bp, float* __restrict__ kscan)
{
    const int row = blockIdx.x, tid = threadIdx.x, c0 = tid * 4, h = tid >> 4;
    __shared__ float ps[256];
    __shared__ float hn[16];
    const size_t base = (size_t)row * C_DIM;
    float kk[4], s = 0.f;
#pragma unroll
    for (int i = 0; i < 4; ++i) { kk[i] = kraw[base + c0 + i] * kkw[c0 + i]; s += kk[i] * kk[i]; }
    ps[tid] = s;
    __syncthreads();
    if ((tid & 15) == 0) {
        float S = 0.f;
        for (int j = 0; j < 16; ++j) S += ps[tid + j];
        hn[h] = 1.f / fmaxf(sqrtf(S), 1e-12f);
    }
    __syncthreads();
    const float inv = hn[h];
#pragma unroll
    for (int i = 0; i < 4; ++i) {
        const int c = c0 + i;
        const float kkn = kk[i] * inv;
        const float av = a[base + c];
        an[base + c] = -kkn;
        bp[base + c] = kkn * av;
        kscan[base + c] = kraw[base + c] * (1.f + (av - 1.f) * kaw[c]);
    }
}

// ----------------------------- WKV scan ------------------------------------
__global__ __launch_bounds__(256) void k_wkv(
    const float* __restrict__ rg, const float* __restrict__ wdg,
    const float* __restrict__ kg, const float* __restrict__ vg,
    const float* __restrict__ ag, const float* __restrict__ bg,
    float* __restrict__ yg)
{
    const int bh = blockIdx.x;                 // 0..B*H-1
    const int b = bh / H_DIM, h = bh % H_DIM;
    const int tid = threadIdx.x;
    const int row = tid >> 2;                  // state row i, 0..63
    const int seg = tid & 3;
    const int c0 = seg * 16;
    __shared__ float sv[6 * 64];               // r | wd | k | v | a | b
    __shared__ float part[256];
    float st[16];
#pragma unroll
    for (int j = 0; j < 16; ++j) st[j] = 0.f;
    const size_t base0 = ((size_t)b * T_DIM) * C_DIM + (size_t)h * N_DIM;

    for (int t = 0; t < T_DIM; ++t) {
        const size_t base = base0 + (size_t)t * C_DIM;
        if (tid < 64) {
            sv[tid]       = rg[base + tid];
            sv[64 + tid]  = wdg[base + tid];
            sv[128 + tid] = kg[base + tid];
            sv[192 + tid] = vg[base + tid];
            sv[256 + tid] = ag[base + tid];
            sv[320 + tid] = bg[base + tid];
        }
        __syncthreads();
        float p = 0.f;
#pragma unroll
        for (int j = 0; j < 16; ++j) p += st[j] * sv[256 + c0 + j];
        part[tid] = p;
        __syncthreads();
        const float sa = part[row * 4] + part[row * 4 + 1] +
                         part[row * 4 + 2] + part[row * 4 + 3];
        const float vi = sv[192 + row];
        float p2 = 0.f;
#pragma unroll
        for (int j = 0; j < 16; ++j) {
            st[j] = st[j] * sv[64 + c0 + j] + sa * sv[320 + c0 + j] + vi * sv[128 + c0 + j];
            p2 += st[j] * sv[c0 + j];
        }
        __syncthreads();
        part[tid] = p2;
        __syncthreads();
        if (seg == 0)
            yg[base + row] = part[row * 4] + part[row * 4 + 1] +
                             part[row * 4 + 2] + part[row * 4 + 3];
    }
}

// GroupNorm + rk*v + gate -> f16 (input to Wo GEMM)
__global__ __launch_bounds__(256) void k_postscan(
    const float* __restrict__ y, const float* __restrict__ r,
    const float* __restrict__ k2, const float* __restrict__ v,
    const float* __restrict__ g, const float* __restrict__ gnw,
    const float* __restrict__ gnb, const float* __restrict__ rkw,
    _Float16* __restrict__ out16)
{
    const int row = blockIdx.x, tid = threadIdx.x, c0 = tid * 4, h = tid >> 4;
    __shared__ float ps[256], pq[256], pr[256];
    __shared__ float hm[16], hv[16], hr[16];
    const size_t base = (size_t)row * C_DIM;
    float yv[4], s = 0.f, q = 0.f, rp = 0.f;
#pragma unroll
    for (int i = 0; i < 4; ++i) {
        const int c = c0 + i;
        yv[i] = y[base + c];
        s += yv[i]; q += yv[i] * yv[i];
        rp += r[base + c] * k2[base + c] * rkw[c];
    }
    ps[tid] = s; pq[tid] = q; pr[tid] = rp;
    __syncthreads();
    if ((tid & 15) == 0) {
        float S = 0.f, Q = 0.f, R = 0.f;
        for (int j = 0; j < 16; ++j) { S += ps[tid + j]; Q += pq[tid + j]; R += pr[tid + j]; }
        const float m = S * (1.f / N_DIM);
        hm[h] = m;
        hv[h] = rsqrtf(Q * (1.f / N_DIM) - m * m + GN_EPS);
        hr[h] = R;
    }
    __syncthreads();
    const float mean = hm[h], inv = hv[h], rk = hr[h];
#pragma unroll
    for (int i = 0; i < 4; ++i) {
        const int c = c0 + i;
        float o = (yv[i] - mean) * inv * gnw[c] + gnb[c];
        o += rk * v[base + c];
        out16[base + c] = (_Float16)(o * g[base + c]);
    }
}

// ----------------------------- host side -----------------------------------
extern "C" void kernel_launch(void* const* d_in, const int* in_sizes, int n_in,
                              void* d_out, int out_size, void* d_ws, size_t ws_size,
                              hipStream_t stream)
{
    (void)in_sizes; (void)n_in; (void)out_size; (void)ws_size;
    auto F = [&](int i) -> const float* { return (const float*)d_in[i]; };
    const int* idx = (const int*)d_in[0];
    // jax tree-flatten (sorted keys): blocks[0..3] (33 tensors each), then
    // emb, head, ln0_b, ln0_w, ln_out_b, ln_out_w
    const int IB = 1;
    const float* emb = F(IB + 4 * 33 + 0);
    const float* head = F(IB + 4 * 33 + 1);
    const float* ln0_b = F(IB + 4 * 33 + 2);
    const float* ln0_w = F(IB + 4 * 33 + 3);
    const float* lnout_b = F(IB + 4 * 33 + 4);
    const float* lnout_w = F(IB + 4 * 33 + 5);

    size_t off = 0;
    auto alloc = [&](size_t bytes) -> void* {
        off = (off + 255) & ~(size_t)255;
        void* p = (void*)((char*)d_ws + off);
        off += bytes;
        return p;
    };
    const size_t BTC = (size_t)BT * C_DIM;

    // f16 weights: transposed [Npad,K], zero-padded
    _Float16 *Wr16[4], *Wk16[4], *Wv16[4], *Wo16[4], *w116[4], *w216[4],
             *a116[4], *a216[4], *v116[4], *v216[4], *g116[4], *g216[4],
             *fk16[4], *fv16[4], *head16;
    auto cvtT = [&](const float* src, int K, int N) -> _Float16* {
        const int Npad = (N + 127) & ~127;
        _Float16* p = (_Float16*)alloc((size_t)Npad * K * sizeof(_Float16));
        dim3 g(K / 32, Npad / 32);
        k_cvtT16<<<g, dim3(32, 8), 0, stream>>>(src, p, K, N, Npad);
        return p;
    };
    for (int l = 0; l < 4; ++l) {
        const int ib = IB + l * 33;
        Wk16[l] = cvtT(F(ib + 0), C_DIM, C_DIM);
        Wo16[l] = cvtT(F(ib + 1), C_DIM, C_DIM);
        Wr16[l] = cvtT(F(ib + 2), C_DIM, C_DIM);
        Wv16[l] = cvtT(F(ib + 3), C_DIM, C_DIM);
        a116[l] = cvtT(F(ib + 5), C_DIM, 64);
        a216[l] = cvtT(F(ib + 6), 64, C_DIM);
        fk16[l] = cvtT(F(ib + 7), C_DIM, F_DIM);
        fv16[l] = cvtT(F(ib + 8), F_DIM, C_DIM);
        g116[l] = cvtT(F(ib + 10), C_DIM, 160);
        g216[l] = cvtT(F(ib + 11), 160, C_DIM);
        v116[l] = cvtT(F(ib + 22), C_DIM, 32);
        v216[l] = cvtT(F(ib + 23), 32, C_DIM);
        w116[l] = cvtT(F(ib + 25), C_DIM, 64);
        w216[l] = cvtT(F(ib + 26), 64, C_DIM);
    }
    head16 = cvtT(head, C_DIM, V_DIM);

    // activations
    float* x      = (float*)alloc(BTC * 4);
    float* xl     = (float*)alloc(BTC * 4);
    float* vfirst = (float*)alloc(BTC * 4);
    float* rbuf   = (float*)alloc(BTC * 4);
    float* wdbuf  = (float*)alloc(BTC * 4);
    float* kraw   = (float*)alloc(BTC * 4);
    float* vv     = (float*)alloc(BTC * 4);
    float* abuf   = (float*)alloc(BTC * 4);
    float* gbuf   = (float*)alloc(BTC * 4);
    float* anb    = (float*)alloc(BTC * 4);
    float* bpb    = (float*)alloc(BTC * 4);
    float* kscan  = (float*)alloc(BTC * 4);
    float* ybuf   = (float*)alloc(BTC * 4);
    float* tmp    = (float*)alloc(BTC * 4);
    float* low    = (float*)alloc((size_t)BT * 192 * 4);
    float* hid    = (float*)alloc((size_t)BT * F_DIM * 4);
    _Float16* low16 = (_Float16*)alloc((size_t)BT * 192 * 2);
    _Float16* hid16 = (_Float16*)alloc((size_t)BT * F_DIM * 2);
    _Float16 *xr16 = (_Float16*)alloc(BTC * 2), *xw16 = (_Float16*)alloc(BTC * 2),
             *xk16 = (_Float16*)alloc(BTC * 2), *xv16 = (_Float16*)alloc(BTC * 2),
             *xa16 = (_Float16*)alloc(BTC * 2), *xg16 = (_Float16*)alloc(BTC * 2);
    _Float16* o16  = (_Float16*)alloc(BTC * 2);
    _Float16* kx16 = (_Float16*)alloc(BTC * 2);
    _Float16* x16  = (_Float16*)alloc(BTC * 2);

    auto gemm = [&](const _Float16* A, const _Float16* Bw, float* Co,
                    int M, int N, int K) {
        dim3 g((N + BN - 1) / BN, (M + BM - 1) / BM);
        k_gemm_f16<<<g, 256, 0, stream>>>(A, Bw, Co, M, N, K);
    };
    const int EWB = (int)((BTC + 255) / 256);

    // embed + ln0
    k_embed<<<EWB, 256, 0, stream>>>(idx, emb, tmp);
    k_layernorm<<<BT, 256, 0, stream>>>(tmp, ln0_w, ln0_b, x, nullptr);

    for (int l = 0; l < 4; ++l) {
        const int ib = IB + l * 33;
        const float *a0 = F(ib + 4), *ffn_xk = F(ib + 9), *gn_b = F(ib + 12),
                    *gn_w = F(ib + 13), *k_a = F(ib + 14), *k_k = F(ib + 15),
                    *ln1_b = F(ib + 16), *ln1_w = F(ib + 17), *ln2_b = F(ib + 18),
                    *ln2_w = F(ib + 19), *r_k = F(ib + 20), *v0 = F(ib + 21),
                    *w0 = F(ib + 24), *x_a = F(ib + 27), *x_g = F(ib + 28),
                    *x_k = F(ib + 29), *x_r = F(ib + 30), *x_v = F(ib + 31),
                    *x_w = F(ib + 32);

        // ---- tmix ----
        k_layernorm<<<BT, 256, 0, stream>>>(x, ln1_w, ln1_b, xl, nullptr);
        k_tmix_pre<<<EWB, 256, 0, stream>>>(xl, x_r, x_w, x_k, x_v, x_a, x_g,
                                            xr16, xw16, xk16, xv16, xa16, xg16);
        gemm(xr16, Wr16[l], rbuf, BT, C_DIM, C_DIM);
        gemm(xw16, w116[l], low, BT, 64, C_DIM);
        k_tanh16<<<(BT * 64 + 255) / 256, 256, 0, stream>>>(low, low16, (size_t)BT * 64);
        gemm(low16, w216[l], tmp, BT, C_DIM, 64);
        k_wact<<<EWB, 256, 0, stream>>>(tmp, w0, wdbuf, BTC);
        gemm(xk16, Wk16[l], kraw, BT, C_DIM, C_DIM);
        gemm(xv16, Wv16[l], vv, BT, C_DIM, C_DIM);
        if (l == 0) {
            hipMemcpyAsync(vfirst, vv, BTC * 4, hipMemcpyDeviceToDevice, stream);
        } else {
            gemm(xv16, v116[l], low, BT, 32, C_DIM);
            k_cvt16<<<(BT * 32 + 255) / 256, 256, 0, stream>>>(low, low16, (size_t)BT * 32);
            gemm(low16, v216[l], tmp, BT, C_DIM, 32);
            k_vmix<<<EWB, 256, 0, stream>>>(vv, vfirst, tmp, v0, BTC);
        }
        gemm(xa16, a116[l], low, BT, 64, C_DIM);
        k_cvt16<<<(BT * 64 + 255) / 256, 256, 0, stream>>>(low, low16, (size_t)BT * 64);
        gemm(low16, a216[l], tmp, BT, C_DIM, 64);
        k_aact<<<EWB, 256, 0, stream>>>(tmp, a0, abuf, BTC);
        gemm(xg16, g116[l], low, BT, 160, C_DIM);
        k_sig16<<<(BT * 160 + 255) / 256, 256, 0, stream>>>(low, low16, (size_t)BT * 160);
        gemm(low16, g216[l], gbuf, BT, C_DIM, 160);

        k_kkprep<<<BT, 256, 0, stream>>>(kraw, abuf, k_k, k_a, anb, bpb, kscan);
        k_wkv<<<B_DIM * H_DIM, 256, 0, stream>>>(rbuf, wdbuf, kscan, vv, anb, bpb, ybuf);
        k_postscan<<<BT, 256, 0, stream>>>(ybuf, rbuf, kscan, vv, gbuf,
                                           gn_w, gn_b, r_k, o16);
        gemm(o16, Wo16[l], tmp, BT, C_DIM, C_DIM);
        k_add<<<EWB, 256, 0, stream>>>(x, tmp, BTC);

        // ---- cmix ----
        k_layernorm<<<BT, 256, 0, stream>>>(x, ln2_w, ln2_b, xl, nullptr);
        k_cmix_pre<<<EWB, 256, 0, stream>>>(xl, ffn_xk, kx16);
        gemm(kx16, fk16[l], hid, BT, F_DIM, C_DIM);
        k_relu2_16<<<(int)(((size_t)BT * F_DIM + 255) / 256), 256, 0, stream>>>(
            hid, hid16, (size_t)BT * F_DIM);
        gemm(hid16, fv16[l], tmp, BT, C_DIM, F_DIM);
        k_add<<<EWB, 256, 0, stream>>>(x, tmp, BTC);
    }

    // final LN + head
    k_layernorm<<<BT, 256, 0, stream>>>(x, lnout_w, lnout_b, nullptr, x16);
    gemm(x16, head16, (float*)d_out, BT, V_DIM, C_DIM);
}
